// SRFTSketchOp_73169062854743
// MI455X (gfx1250) — compile-verified, matrix operand order checked by reading, not run on hardware
//
#include <hip/hip_runtime.h>

typedef __attribute__((ext_vector_type(16))) _Float16 v16h;
typedef __attribute__((ext_vector_type(2)))  _Float16 v2h;
typedef __attribute__((ext_vector_type(8)))  float    v8f;
typedef __attribute__((ext_vector_type(4)))  float    v4f;
typedef __attribute__((ext_vector_type(2)))  float    v2f;

#define N_PARAMS  262144
#define SKETCH_D  4096
#define N_COLS    64
#define NCOL_TOT  128              // BATCH * N_COLS
#define BM        128              // output rows per workgroup
#define BK        32               // K per WMMA step (f16 16x16x32)
#define KSEGS     16               // split-K factor
#define KSEG_LEN  (N_PARAMS / KSEGS)
#define ROW_TILES (SKETCH_D / BM)  // 32
#define MASK4N    0xFFFFFu         // 4N - 1 = 2^20 - 1

__global__ void srft_zero_out(float* __restrict__ out, int n) {
    int i = blockIdx.x * blockDim.x + threadIdx.x;
    if (i < n) out[i] = 0.0f;
}

// O[b, p, l] = (1/N) * [ 0.5*y[0] + sum_{k>=1} y[k] * cos(pi*k*(2*P[p]+1)/(2N)) ]
// with y[k] = D[k]*M[b,k,l].  GEMM: C(4096 x 262144) @ Y(262144 x 128),
// A generated in-register via exact mod-2^20 phase recurrence + v_cos_f32.
__global__ __launch_bounds__(256)
void srft_sketch_wmma(const float* __restrict__ M,
                      const float* __restrict__ D,
                      const int*   __restrict__ P,
                      float* __restrict__ out)
{
    // Double-buffered B tile in WMMA fragment order:
    // [buf][colTile 0..7][lane 0..31][16 halves]
    __shared__ __align__(32) _Float16 ldsB[2][8][32][16];

    const int tid   = threadIdx.x;
    const int lane  = tid & 31;
    const int wv    = tid >> 5;           // wave 0..7 -> 16-row strip
    const int br    = blockIdx.x;         // row tile (0..31)
    const int kbase = (int)blockIdx.y * KSEG_LEN;
    const int kend  = kbase + KSEG_LEN;

    const int hi = lane >> 4;             // lane half selects K sub-block

    // ---- A-operand setup: q = 2*P[row]+1, phase state per fragment half ----
    const int rowIdx = br * BM + wv * 16 + (lane & 15);
    const unsigned q    = 2u * (unsigned)P[rowIdx] + 1u;
    const unsigned step = (32u * q) & MASK4N;      // phase advance per K-step

    unsigned tph[16];                              // (k_j * q) mod 4N
    #pragma unroll
    for (int j = 0; j < 16; ++j) {
        const int kl = ((j >> 3) << 4) + hi * 8 + (j & 7);  // K per ISA A layout
        tph[j] = ((unsigned)(kbase + kl) * q) & MASK4N;
    }

    // ---- staging setup: this thread owns 4 cols x (rows kl0,kl0+1,kl0+16,kl0+17)
    const int col4 = (tid & 31) * 4;               // 0..124, step 4
    const int kl0  = (tid >> 5) * 2;               // 0,2,..,14
    const int bcol = col4 >> 6;
    const int lcol = col4 & 63;
    const float* pM = M + (size_t)bcol * ((size_t)N_PARAMS * N_COLS)
                        + (size_t)(kbase + kl0) * N_COLS + lcol;
    const float* pD = D + (kbase + kl0);

    // fragment-order LDS destination (loop-invariant):
    // c = col>>4, lane = hig*16 + (col&15), j = g*8 + (kl&7); kl pair -> one dword
    const int c0   = col4 >> 4;
    const int hig  = (kl0 >> 3) & 1;
    const int off0 = (((c0 * 32) + hig * 16 + (col4 & 15)) * 16 + (kl0 & 7)) * 2;

    v8f acc[8] = {};                               // 16 x 128 f32 per wave
    int par = 0;

    for (int k0 = kbase; k0 < kend; k0 += BK) {
        // ---- prefetch next K chunk (global_prefetch_b8) ----
        if (k0 + BK < kend) {
            __builtin_prefetch(pM + BK * N_COLS, 0, 1);
            __builtin_prefetch(pM + BK * N_COLS + 16 * N_COLS, 0, 1);
        }

        // ---- stage Y tile: 4x b128 loads, *D, pack pairs, 8x b32 LDS stores ----
        const v4f a0 = *(const v4f*)(pM);                       // row kl0
        const v4f a1 = *(const v4f*)(pM + N_COLS);              // row kl0+1
        const v4f b0 = *(const v4f*)(pM + 16 * N_COLS);         // row kl0+16
        const v4f b1 = *(const v4f*)(pM + 17 * N_COLS);         // row kl0+17
        const v2f d0 = *(const v2f*)(pD);                       // D[kl0], D[kl0+1]
        const v2f d1 = *(const v2f*)(pD + 16);                  // D[kl0+16..17]

        char* sb = (char*)(&ldsB[par][0][0][0]) + off0;
        #pragma unroll
        for (int cc = 0; cc < 4; ++cc) {
            v2h p0, p1;
            p0.x = (_Float16)(a0[cc] * d0.x);
            p0.y = (_Float16)(a1[cc] * d0.y);
            p1.x = (_Float16)(b0[cc] * d1.x);
            p1.y = (_Float16)(b1[cc] * d1.y);
            *(v2h*)(sb + cc * 32)      = p0;   // g=0 halves (j = kl&7, +1)
            *(v2h*)(sb + cc * 32 + 16) = p1;   // g=1 halves (j = 8+(kl&7), +1)
        }
        pM += BK * N_COLS;
        pD += BK;

        // ---- generate A fragment: v_cos_f32 on carried phase state ----
        v16h afrag;
        #pragma unroll
        for (int j = 0; j < 16; ++j) {
            // v_cos_f32 computes cos(2*pi*x); x = t/2^20 -> cos(pi*k*(2p+1)/(2N))
            afrag[j] = (_Float16)__builtin_amdgcn_cosf(
                           (float)tph[j] * 9.5367431640625e-7f);
            tph[j] = (tph[j] + step) & MASK4N;
        }
        if (k0 == 0 && hi == 0)                    // DC term weight 0.5 (k==0)
            afrag[0] = (_Float16)((float)afrag[0] * 0.5f);

        __syncthreads();   // single barrier: buffers alternate

        const char* rb = (const char*)ldsB + par * 8192 + lane * 32;
        #pragma unroll
        for (int c = 0; c < 8; ++c) {
            const v16h bfrag = *(const v16h*)(rb + c * 1024);
            acc[c] = __builtin_amdgcn_wmma_f32_16x16x32_f16(
                false, afrag, false, bfrag, (short)0, acc[c], false, false);
        }
        par ^= 1;
    }

    // ---- epilogue: scale by 1/N, split-K accumulate ----
    const float scale = 1.0f / (float)N_PARAMS;
    #pragma unroll
    for (int c = 0; c < 8; ++c) {
        const int col = c * 16 + (lane & 15);      // C/D layout: N = lane&15
        const int bo  = col >> 6;
        const int lo  = col & 63;
        #pragma unroll
        for (int r = 0; r < 8; ++r) {
            const int mo = r + hi * 8;             // C/D layout: M = r + 8*hi
            const int ro = br * BM + wv * 16 + mo;
            atomicAdd(&out[(size_t)bo * ((size_t)SKETCH_D * N_COLS)
                           + (size_t)ro * N_COLS + lo],
                      acc[c][r] * scale);
        }
    }
}

extern "C" void kernel_launch(void* const* d_in, const int* in_sizes, int n_in,
                              void* d_out, int out_size, void* d_ws, size_t ws_size,
                              hipStream_t stream) {
    (void)in_sizes; (void)n_in; (void)d_ws; (void)ws_size;
    const float* M = (const float*)d_in[0];
    const float* D = (const float*)d_in[1];
    const int*   P = (const int*)d_in[2];
    float* out = (float*)d_out;

    srft_zero_out<<<(out_size + 255) / 256, 256, 0, stream>>>(out, out_size);

    dim3 grid(ROW_TILES, KSEGS);  // 32 row tiles x 16 K segments = 512 WGs
    srft_sketch_wmma<<<grid, 256, 0, stream>>>(M, D, P, out);
}
